// Decoder_44530220925559
// MI455X (gfx1250) — matrix operand order repaired and checked by reading, not compiled, not run on hardware
//
#include <hip/hip_runtime.h>
#include <math.h>

// ---------------------------------------------------------------------------
// Problem dims (from reference): T=64, B=32, S=128, H=512, E=256, V=32000
// ---------------------------------------------------------------------------
#define TT 64
#define BB 32
#define SS 128
#define HH 512
#define EE 256
#define VV 32000

typedef __attribute__((ext_vector_type(16))) __bf16 v16bf;
typedef __attribute__((ext_vector_type(8)))  float  v8f;
typedef unsigned short bfraw;   // bf16 storage type (host-safe)

// ---------------- bf16 helpers (storage = ushort, RNE convert) -------------
__device__ __forceinline__ bfraw f2bf(float x) {
  union { float f; unsigned u; } t; t.f = x;
  return (bfraw)((t.u + 0x7FFFu + ((t.u >> 16) & 1u)) >> 16);
}
__device__ __forceinline__ float sigf(float x) { return 1.0f / (1.0f + __expf(-x)); }

// ---------------------------------------------------------------------------
// CDNA5 WMMA 16x16x32 bf16 fragment loaders (wave32; contiguous k = 16*(lane/16) + s in A and B, not the ISA table order).
//
// A (16x32, MxK), row-major src with leading dim lda:
//   lanes 0-15 : row M=lane,   K=0..15
//   lanes 16-31: row M=lane-16,K=16..31
// ---------------------------------------------------------------------------
__device__ __forceinline__ v16bf load_a_bf(const bfraw* A, int lda, int m0, int k0) {
  int lane = threadIdx.x & 31;
  const bfraw* p = A + (size_t)(m0 + (lane & 15)) * lda + (k0 + ((lane >> 4) << 4));
  union { v16bf v; bfraw s[16]; } t;
#pragma unroll
  for (int i = 0; i < 8; ++i) t.s[i] = p[i];          // K = kh+0..7
#pragma unroll
  for (int i = 0; i < 8; ++i) t.s[8 + i] = p[8 + i];  // K = kh+8..15
  return t.v;
}

// B (32x16, KxN). Source W is (N,K) row-major => column n of B = row n of W.
//   lanes 0-15 : col N=lane,    K=0..15  (16 contiguous bf16)
//   lanes 16-31: col N=lane-16, K=16..31
__device__ __forceinline__ v16bf load_b_bf(const bfraw* W, int ldk, int n0, int k0) {
  int lane = threadIdx.x & 31;
  const bfraw* p = W + (size_t)(n0 + (lane & 15)) * ldk + (k0 + ((lane >> 4) << 4));
  union { v16bf v; uint4 u[2]; } t;
  t.u[0] = ((const uint4*)p)[0];
  t.u[1] = ((const uint4*)p)[1];
  return t.v;
}

__device__ __forceinline__ v8f wmma_bf(v16bf a, v16bf b, v8f c) {
  // v_wmma_f32_16x16x32_bf16  D = A*B + C   (fp32 accumulate)
  return __builtin_amdgcn_wmma_f32_16x16x32_bf16(false, a, false, b, (short)0, c,
                                                 false, false);
}

// C/D layout: lane l -> col n0+(l&15); vgpr i -> row m0 + i + 8*(l>>4)

// ---------------------------------------------------------------------------
// One-time prep kernels
// ---------------------------------------------------------------------------
__global__ __launch_bounds__(256) void k_f2bf(const float* __restrict__ s,
                                              bfraw* __restrict__ d, int n) {
  for (int i = blockIdx.x * blockDim.x + threadIdx.x; i < n; i += gridDim.x * blockDim.x)
    d[i] = f2bf(s[i]);
}

__global__ __launch_bounds__(256) void k_gather_embs(const int* __restrict__ tgt_in,
                                                     const float* __restrict__ embed,
                                                     bfraw* __restrict__ embs) {
  int n = TT * BB * EE;
  for (int i = blockIdx.x * blockDim.x + threadIdx.x; i < n; i += gridDim.x * blockDim.x) {
    int e = i % EE, tb = i / EE;
    embs[i] = f2bf(embed[(size_t)tgt_in[tb] * EE + e]);
  }
}

__global__ __launch_bounds__(256) void k_init_state(const float* __restrict__ h0,
                                                    const float* __restrict__ c0,
                                                    float* h, float* c,
                                                    bfraw* h_bf, bfraw* att_bf) {
  int n = BB * HH;
  for (int i = blockIdx.x * blockDim.x + threadIdx.x; i < n; i += gridDim.x * blockDim.x) {
    float hv = h0[i];
    h[i] = hv; c[i] = c0[i]; h_bf[i] = f2bf(hv); att_bf[i] = 0; // bf16 zero
  }
}

// ---------------------------------------------------------------------------
// src_proj = src_enc(4096x512) @ W_attn_in^T(512x512) -> fp32 (4096x512)
// One 16x16 tile per wave, K=512 -> 16 wmma per tile.
// ---------------------------------------------------------------------------
__global__ __launch_bounds__(256) void k_srcproj(const bfraw* __restrict__ A,
                                                 const bfraw* __restrict__ W,
                                                 float* __restrict__ out) {
  int gwave = blockIdx.x * (blockDim.x >> 5) + (threadIdx.x >> 5);
  int mt = gwave >> 5;        // 256 M-tiles (M=4096)
  int nt = gwave & 31;        // 32  N-tiles (N=512)
  int m0 = mt * 16, n0 = nt * 16;
  v8f acc = {};
#pragma unroll 4
  for (int k0 = 0; k0 < HH; k0 += 32)
    acc = wmma_bf(load_a_bf(A, HH, m0, k0), load_b_bf(W, HH, n0, k0), acc);
  int lane = threadIdx.x & 31, cn = lane & 15, rb = (lane >> 4) * 8;
#pragma unroll
  for (int i = 0; i < 8; ++i)
    out[(size_t)(m0 + rb + i) * HH + n0 + cn] = acc[i];
}

// ---------------------------------------------------------------------------
// One decode timestep, single 1024-thread workgroup (32 waves), all phases.
// ---------------------------------------------------------------------------
__global__ __launch_bounds__(1024) void k_step(
    int t,
    const bfraw* __restrict__ embs_bf,   // (T,B,E)
    const bfraw* __restrict__ Wih_bf,    // (4H, E+H)
    const bfraw* __restrict__ Whh_bf,    // (4H, H)
    const float* __restrict__ b_ih, const float* __restrict__ b_hh,
    float* __restrict__ h_f, float* __restrict__ c_f,
    bfraw* __restrict__ h_bf, bfraw* __restrict__ att_bf,
    bfraw* __restrict__ ctx_bf,
    float* __restrict__ gates,           // (B, 4H)
    const float* __restrict__ srcproj,   // (B*S, H) fp32
    const float* __restrict__ src_enc,   // (B,S,H)  fp32
    const int* __restrict__ src_lens,
    const bfraw* __restrict__ Wao_bf,    // (H, 2H)
    const bfraw* __restrict__ Wgen_bf,   // (H, E+2H)
    const float* __restrict__ b_gen,
    bfraw* __restrict__ decs_bf)         // (T*B, H)
{
  __shared__ float sc[BB][SS];           // scores -> alpha (16 KB)

  const int wave = threadIdx.x >> 5, lane = threadIdx.x & 31;
  const int cn = lane & 15, rb = (lane >> 4) * 8;
  const bfraw* embT = embs_bf + (size_t)t * BB * EE;   // (32,256)

  // ---- Phase 1: gates = [emb|att] @ W_ih^T + h @ W_hh^T + b  (M=32,N=2048)
#pragma unroll
  for (int j = 0; j < 8; ++j) {
    int tid = wave * 8 + j;              // 256 tiles: 2 M x 128 N
    int mt = tid & 1, nt = tid >> 1;
    int m0 = mt * 16, n0 = nt * 16;
    float bias = b_ih[n0 + cn] + b_hh[n0 + cn];
    v8f acc;
#pragma unroll
    for (int i = 0; i < 8; ++i) acc[i] = bias;
    for (int k0 = 0; k0 < EE; k0 += 32)  // emb part (K=256)
      acc = wmma_bf(load_a_bf(embT, EE, m0, k0),
                    load_b_bf(Wih_bf, EE + HH, n0, k0), acc);
    for (int k0 = 0; k0 < HH; k0 += 32)  // att (input feed) part (K=512)
      acc = wmma_bf(load_a_bf(att_bf, HH, m0, k0),
                    load_b_bf(Wih_bf, EE + HH, n0, EE + k0), acc);
    for (int k0 = 0; k0 < HH; k0 += 32)  // recurrent part (K=512)
      acc = wmma_bf(load_a_bf(h_bf, HH, m0, k0),
                    load_b_bf(Whh_bf, HH, n0, k0), acc);
#pragma unroll
    for (int i = 0; i < 8; ++i)
      gates[(size_t)(m0 + rb + i) * (4 * HH) + n0 + cn] = acc[i];
  }
  __threadfence(); __syncthreads();

  // ---- Phase 2: LSTM cell elementwise
  for (int i = threadIdx.x; i < BB * HH; i += blockDim.x) {
    int b = i >> 9, hh = i & (HH - 1);
    const float* g = gates + (size_t)b * (4 * HH);
    float gi = g[hh], gf = g[HH + hh], gg = g[2 * HH + hh], go = g[3 * HH + hh];
    float ct = sigf(gf) * c_f[i] + sigf(gi) * tanhf(gg);
    float ht = sigf(go) * tanhf(ct);
    c_f[i] = ct; h_f[i] = ht; h_bf[i] = f2bf(ht);
  }
  __threadfence(); __syncthreads();

  // ---- Phase 3: scores[b][s] = h_t[b] . src_proj[b,s]  (masked)
  for (int p = threadIdx.x; p < BB * SS; p += blockDim.x) {
    int b = p >> 7, s = p & (SS - 1);
    const float4* hv = (const float4*)(h_f + (size_t)b * HH);
    const float4* sp = (const float4*)(srcproj + (size_t)p * HH);
    float d = 0.f;
#pragma unroll 4
    for (int q = 0; q < HH / 4; ++q) {
      float4 a = hv[q], c = sp[q];
      d += a.x * c.x + a.y * c.y + a.z * c.z + a.w * c.w;
    }
    sc[b][s] = (s < src_lens[b]) ? d : -INFINITY;
  }
  __syncthreads();

  // ---- Phase 4: softmax over S per batch row; one wave per row (32 waves)
  {
    int b = wave;
    float e0 = sc[b][lane], e1 = sc[b][lane + 32],
          e2 = sc[b][lane + 64], e3 = sc[b][lane + 96];
    float mx = fmaxf(fmaxf(e0, e1), fmaxf(e2, e3));
#pragma unroll
    for (int m = 16; m >= 1; m >>= 1) mx = fmaxf(mx, __shfl_xor(mx, m, 32));
    float x0 = __expf(e0 - mx), x1 = __expf(e1 - mx),
          x2 = __expf(e2 - mx), x3 = __expf(e3 - mx);
    float s = x0 + x1 + x2 + x3;
#pragma unroll
    for (int m = 16; m >= 1; m >>= 1) s += __shfl_xor(s, m, 32);
    float inv = 1.f / s;
    sc[b][lane] = x0 * inv; sc[b][lane + 32] = x1 * inv;
    sc[b][lane + 64] = x2 * inv; sc[b][lane + 96] = x3 * inv;
  }
  __syncthreads();

  // ---- Phase 5: ctx[b][h] = sum_s alpha[b][s] * src_enc[b,s,h]
  for (int i = threadIdx.x; i < BB * HH; i += blockDim.x) {
    int b = i >> 9, hh = i & (HH - 1);
    const float* se = src_enc + ((size_t)b * SS) * HH + hh;
    float acc = 0.f;
#pragma unroll 4
    for (int s = 0; s < SS; ++s) acc += sc[b][s] * se[(size_t)s * HH];
    ctx_bf[i] = f2bf(acc);
  }
  __threadfence(); __syncthreads();

  // ---- Phase 6: att_t = tanh([ctx, h_t] @ W_attn_out^T)  (M=32,N=512,K=1024)
#pragma unroll
  for (int j = 0; j < 2; ++j) {
    int tid = wave * 2 + j;              // 64 tiles
    int mt = tid & 1, nt = tid >> 1;
    int m0 = mt * 16, n0 = nt * 16;
    v8f acc = {};
    for (int k0 = 0; k0 < HH; k0 += 32)
      acc = wmma_bf(load_a_bf(ctx_bf, HH, m0, k0),
                    load_b_bf(Wao_bf, 2 * HH, n0, k0), acc);
    for (int k0 = 0; k0 < HH; k0 += 32)
      acc = wmma_bf(load_a_bf(h_bf, HH, m0, k0),
                    load_b_bf(Wao_bf, 2 * HH, n0, HH + k0), acc);
#pragma unroll
    for (int i = 0; i < 8; ++i)
      att_bf[(size_t)(m0 + rb + i) * HH + n0 + cn] = f2bf(tanhf(acc[i]));
  }
  __threadfence(); __syncthreads();

  // ---- Phase 7: dec = tanh([emb, h_t, att_t] @ W_gen^T + b_gen)
#pragma unroll
  for (int j = 0; j < 2; ++j) {
    int tid = wave * 2 + j;
    int mt = tid & 1, nt = tid >> 1;
    int m0 = mt * 16, n0 = nt * 16;
    float bias = b_gen[n0 + cn];
    v8f acc;
#pragma unroll
    for (int i = 0; i < 8; ++i) acc[i] = bias;
    for (int k0 = 0; k0 < EE; k0 += 32)
      acc = wmma_bf(load_a_bf(embT, EE, m0, k0),
                    load_b_bf(Wgen_bf, EE + 2 * HH, n0, k0), acc);
    for (int k0 = 0; k0 < HH; k0 += 32)
      acc = wmma_bf(load_a_bf(h_bf, HH, m0, k0),
                    load_b_bf(Wgen_bf, EE + 2 * HH, n0, EE + k0), acc);
    for (int k0 = 0; k0 < HH; k0 += 32)
      acc = wmma_bf(load_a_bf(att_bf, HH, m0, k0),
                    load_b_bf(Wgen_bf, EE + 2 * HH, n0, EE + HH + k0), acc);
#pragma unroll
    for (int i = 0; i < 8; ++i)
      decs_bf[((size_t)t * BB + m0 + rb + i) * HH + n0 + cn] = f2bf(tanhf(acc[i]));
  }
}

// ---------------------------------------------------------------------------
// Fused readout: logits = decs @ W_read^T + b_read, streaming online
// log-softmax + argmax + picked-logit per row (never materializes logits).
// Block = 16 rows. The 16 A-fragments (16x512 bf16 = 128 VGPRs) are hoisted
// into registers for the whole V sweep; the inner loop is then pure
// B-stream (global b128, L2-resident) + 16 WMMA + online-softmax VALU.
// ---------------------------------------------------------------------------
__global__ __launch_bounds__(256) void k_readout(const bfraw* __restrict__ decs,
                                                 const bfraw* __restrict__ Wread,
                                                 const float* __restrict__ b_read,
                                                 const int* __restrict__ tgt_out,
                                                 float* __restrict__ loss_out,
                                                 float* __restrict__ ys_out) {
  __shared__ bfraw sA[16 * HH];                // 16 KB A tile (coalesced stage)
  __shared__ float red_m[8][16], red_s[8][16], red_v[8][16], red_p[8][16];
  __shared__ int   red_i[8][16];

  const int r0 = blockIdx.x * 16;
  for (int i = threadIdx.x; i < 16 * HH; i += blockDim.x)
    sA[i] = decs[(size_t)r0 * HH + i];
  __syncthreads();

  const int wave = threadIdx.x >> 5, lane = threadIdx.x & 31;
  const int cn = lane & 15, rb = (lane >> 4) * 8;

  // Loop-invariant A fragments: extract once from LDS into registers.
  v16bf afr[HH / 32];
#pragma unroll
  for (int kf = 0; kf < HH / 32; ++kf)
    afr[kf] = load_a_bf(sA, HH, 0, kf * 32);

  float mx[8], sm[8], amv[8], pk[8]; int ami[8], tgt[8];
#pragma unroll
  for (int i = 0; i < 8; ++i) {
    mx[i] = -INFINITY; sm[i] = 0.f; amv[i] = -INFINITY; ami[i] = 0; pk[i] = 0.f;
    tgt[i] = tgt_out[r0 + rb + i];
  }

  for (int nc = 0; nc < VV; nc += 128) {       // 250 chunks of 128 vocab cols
    int n0 = nc + wave * 16;
    __builtin_prefetch(Wread + (size_t)(n0 + 128) * HH, 0, 1);
    v8f acc = {};
#pragma unroll 4
    for (int kf = 0; kf < HH / 32; ++kf)
      acc = wmma_bf(afr[kf], load_b_bf(Wread, HH, n0, kf * 32), acc);
    int n = n0 + cn;
    float br = b_read[n];
#pragma unroll
    for (int i = 0; i < 8; ++i) {
      float val = acc[i] + br;
      float nm = fmaxf(mx[i], val);
      sm[i] = sm[i] * __expf(mx[i] - nm) + __expf(val - nm);
      mx[i] = nm;
      if (val > amv[i]) { amv[i] = val; ami[i] = n; }
      if (n == tgt[i]) pk[i] = val;            // exactly one hit per row
    }
  }

  // cross-lane merge over the 16 column-lanes (xor bits 0..3 keep lane-half)
#pragma unroll
  for (int m = 1; m < 16; m <<= 1) {
#pragma unroll
    for (int i = 0; i < 8; ++i) {
      float om = __shfl_xor(mx[i], m, 32);
      float os = __shfl_xor(sm[i], m, 32);
      float nm = fmaxf(mx[i], om);
      sm[i] = sm[i] * __expf(mx[i] - nm) + os * __expf(om - nm);
      mx[i] = nm;
      float ov = __shfl_xor(amv[i], m, 32);
      int   oi = __shfl_xor(ami[i], m, 32);
      if (ov > amv[i] || (ov == amv[i] && oi < ami[i])) { amv[i] = ov; ami[i] = oi; }
      pk[i] += __shfl_xor(pk[i], m, 32);
    }
  }
  if ((lane & 15) == 0) {
#pragma unroll
    for (int i = 0; i < 8; ++i) {
      int r = rb + i;
      red_m[wave][r] = mx[i]; red_s[wave][r] = sm[i];
      red_v[wave][r] = amv[i]; red_i[wave][r] = ami[i]; red_p[wave][r] = pk[i];
    }
  }
  __syncthreads();

  if (threadIdx.x < 16) {                      // final cross-wave merge per row
    int r = threadIdx.x;
    float M = red_m[0][r], S = red_s[0][r], AV = red_v[0][r], P = red_p[0][r];
    int AI = red_i[0][r];
    for (int w = 1; w < 8; ++w) {
      float om = red_m[w][r], os = red_s[w][r];
      float nm = fmaxf(M, om);
      S = S * __expf(M - nm) + os * __expf(om - nm);
      M = nm;
      float ov = red_v[w][r]; int oi = red_i[w][r];
      if (ov > AV || (ov == AV && oi < AI)) { AV = ov; AI = oi; }
      P += red_p[w][r];
    }
    float logZ = M + logf(S);
    int row = r0 + r;
    float msk = (tgt_out[row] != 0) ? 1.f : 0.f;  // PAD == 0
    loss_out[row] = -(P - logZ) * msk;
    ys_out[row] = (float)AI;
  }
}

// ---------------------------------------------------------------------------
// Host launcher
// ---------------------------------------------------------------------------
extern "C" void kernel_launch(void* const* d_in, const int* in_sizes, int n_in,
                              void* d_out, int out_size, void* d_ws, size_t ws_size,
                              hipStream_t stream) {
  (void)in_sizes; (void)n_in; (void)out_size; (void)ws_size;
  const int*   tgt_in   = (const int*)d_in[0];
  const int*   tgt_out  = (const int*)d_in[1];
  const float* src_enc  = (const float*)d_in[2];
  const int*   src_lens = (const int*)d_in[3];
  const float* h0       = (const float*)d_in[4];
  const float* c0       = (const float*)d_in[5];
  const float* embed    = (const float*)d_in[6];
  const float* W_ih     = (const float*)d_in[7];
  const float* b_ih     = (const float*)d_in[8];
  const float* W_hh     = (const float*)d_in[9];
  const float* b_hh     = (const float*)d_in[10];
  const float* W_ai     = (const float*)d_in[11];
  const float* W_ao     = (const float*)d_in[12];
  const float* W_gen    = (const float*)d_in[13];
  const float* b_gen    = (const float*)d_in[14];
  const float* W_read   = (const float*)d_in[15];
  const float* b_read   = (const float*)d_in[16];

  char* base = (char*)d_ws;
  size_t off = 0;
  auto alloc = [&](size_t bytes) -> void* {
    void* p = base + off;
    off = (off + bytes + 255) & ~(size_t)255;
    return p;
  };
  bfraw* embs_bf  = (bfraw*)alloc((size_t)TT * BB * EE * 2);
  bfraw* senc_bf  = (bfraw*)alloc((size_t)BB * SS * HH * 2);
  bfraw* Wih_bf   = (bfraw*)alloc((size_t)4 * HH * (EE + HH) * 2);
  bfraw* Whh_bf   = (bfraw*)alloc((size_t)4 * HH * HH * 2);
  bfraw* Wai_bf   = (bfraw*)alloc((size_t)HH * HH * 2);
  bfraw* Wao_bf   = (bfraw*)alloc((size_t)HH * 2 * HH * 2);
  bfraw* Wgen_bf  = (bfraw*)alloc((size_t)HH * (EE + 2 * HH) * 2);
  bfraw* Wread_bf = (bfraw*)alloc((size_t)VV * HH * 2);
  float* srcproj  = (float*)alloc((size_t)BB * SS * HH * 4);
  float* gates    = (float*)alloc((size_t)BB * 4 * HH * 4);
  float* h_f      = (float*)alloc((size_t)BB * HH * 4);
  float* c_f      = (float*)alloc((size_t)BB * HH * 4);
  bfraw* h_bf     = (bfraw*)alloc((size_t)BB * HH * 2);
  bfraw* att_bf   = (bfraw*)alloc((size_t)BB * HH * 2);
  bfraw* ctx_bf   = (bfraw*)alloc((size_t)BB * HH * 2);
  bfraw* decs_bf  = (bfraw*)alloc((size_t)TT * BB * HH * 2);

  auto cvt = [&](const float* s, bfraw* d, size_t n) {
    k_f2bf<<<dim3((unsigned)((n + 255) / 256)), dim3(256), 0, stream>>>(s, d, (int)n);
  };
  cvt(W_ih,   Wih_bf,   (size_t)4 * HH * (EE + HH));
  cvt(W_hh,   Whh_bf,   (size_t)4 * HH * HH);
  cvt(W_ai,   Wai_bf,   (size_t)HH * HH);
  cvt(W_ao,   Wao_bf,   (size_t)HH * 2 * HH);
  cvt(W_gen,  Wgen_bf,  (size_t)HH * (EE + 2 * HH));
  cvt(W_read, Wread_bf, (size_t)VV * HH);
  cvt(src_enc, senc_bf, (size_t)BB * SS * HH);

  k_gather_embs<<<dim3((TT * BB * EE + 255) / 256), dim3(256), 0, stream>>>(
      tgt_in, embed, embs_bf);
  k_init_state<<<dim3((BB * HH + 255) / 256), dim3(256), 0, stream>>>(
      h0, c0, h_f, c_f, h_bf, att_bf);

  // src_proj GEMM: 8192 wave-tiles, 8 waves/block
  k_srcproj<<<dim3(1024), dim3(256), 0, stream>>>(senc_bf, Wai_bf, srcproj);

  // sequential decode: 64 single-workgroup step kernels
  for (int t = 0; t < TT; ++t)
    k_step<<<dim3(1), dim3(1024), 0, stream>>>(
        t, embs_bf, Wih_bf, Whh_bf, b_ih, b_hh, h_f, c_f, h_bf, att_bf, ctx_bf,
        gates, srcproj, src_enc, src_lens, Wao_bf, Wgen_bf, b_gen, decs_bf);

  // fused readout + online log-softmax/argmax/picked-loss
  float* loss = (float*)d_out;
  float* ys   = loss + TT * BB;
  k_readout<<<dim3(TT * BB / 16), dim3(256), 0, stream>>>(
      decs_bf, Wread_bf, b_read, tgt_out, loss, ys);
}
